// Q6ExpertBank_17428977287672
// MI455X (gfx1250) — matrix operand order repaired and checked by reading, not compiled, not run on hardware
//
#include <hip/hip_runtime.h>
#include <hip/hip_bf16.h>

#define NTOK  2048
#define DDIM  512
#define FF    2048
#define NEXP  64
#define CAP   2048
#define FFC   256     // ff chunk processed per block iteration
#define MT    32      // token tile (two 16-row WMMA sub-tiles)

typedef __attribute__((ext_vector_type(16))) __bf16 v16bf;
typedef __attribute__((ext_vector_type(8)))  float  v8f;

union FragU { uint4 u[2]; v16bf v; };

// A-matrix fragment (16x32 bf16): lane half kh=lane>>4 holds
// K = k0..k0+15 (contiguous, matches B frag), row m = lane&15.
// k0 passed in already includes kh*16.  Two aligned 16B LDS reads.
static __device__ inline v16bf load_a_frag(const __bf16* base, int stride, int row, int k0) {
    FragU f;
    f.u[0] = *(const uint4*)(base + row * stride + k0);
    f.u[1] = *(const uint4*)(base + row * stride + k0 + 8);
    return f.v;
}

// B-matrix fragment (32x16 bf16): lane n = lane&15 is the N column, lane
// half kh selects K = kh*16..kh*16+15 -> 16 consecutive fp32 in the weight
// row.  Load 64B contiguous, convert fp32 -> bf16 in registers.
static __device__ inline v16bf load_b_frag_f32(const float* p) {
    v16bf r;
#pragma unroll
    for (int i = 0; i < 4; ++i) {
        float4 f = ((const float4*)p)[i];
        r[4 * i + 0] = (__bf16)f.x;
        r[4 * i + 1] = (__bf16)f.y;
        r[4 * i + 2] = (__bf16)f.z;
        r[4 * i + 3] = (__bf16)f.w;
    }
    return r;
}

// ---------------- Kernel 1: LayerNorm + top-4 routing ----------------
__global__ __launch_bounds__(256) void ln_topk_kernel(
    const float* __restrict__ x, const float* __restrict__ hexw,
    const float* __restrict__ gamma, const float* __restrict__ beta,
    __bf16* __restrict__ xn, int* __restrict__ ecount,
    int* __restrict__ etok, float* __restrict__ egate)
{
    const int lane = threadIdx.x & 31;
    const int wv   = threadIdx.x >> 5;
    const int tok  = blockIdx.x * 8 + wv;

    const float* xr = x + (size_t)tok * DDIM;
    float v[16];
    float sum = 0.f;
#pragma unroll
    for (int i = 0; i < 16; ++i) { v[i] = xr[i * 32 + lane]; sum += v[i]; }
#pragma unroll
    for (int o = 16; o >= 1; o >>= 1) sum += __shfl_xor(sum, o, 32);
    const float mu = sum * (1.f / DDIM);

    float var = 0.f;
#pragma unroll
    for (int i = 0; i < 16; ++i) { float d = v[i] - mu; var += d * d; }
#pragma unroll
    for (int o = 16; o >= 1; o >>= 1) var += __shfl_xor(var, o, 32);
    const float rs = rsqrtf(var * (1.f / DDIM) + 1e-5f);

#pragma unroll
    for (int i = 0; i < 16; ++i) {
        int col = i * 32 + lane;
        float y = (v[i] - mu) * rs * gamma[col] + beta[col];
        xn[(size_t)tok * DDIM + col] = (__bf16)y;
    }

    // top-4 of 64 expert weights: 2 per lane, shuffle arg-max x4
    float w0 = hexw[(size_t)tok * NEXP + lane];
    float w1 = hexw[(size_t)tok * NEXP + 32 + lane];
    float selv[4]; int seli[4];
#pragma unroll
    for (int t = 0; t < 4; ++t) {
        float bv; int bi;
        if (w0 >= w1) { bv = w0; bi = lane; } else { bv = w1; bi = lane + 32; }
#pragma unroll
        for (int o = 16; o >= 1; o >>= 1) {
            float ov = __shfl_xor(bv, o, 32);
            int   oi = __shfl_xor(bi, o, 32);
            if (ov > bv || (ov == bv && oi < bi)) { bv = ov; bi = oi; }
        }
        selv[t] = bv; seli[t] = bi;
        if (bi == lane)           w0 = -1e30f;
        else if (bi == lane + 32) w1 = -1e30f;
    }
    if (lane == 0) {
        float s = selv[0] + selv[1] + selv[2] + selv[3] + 1e-8f;
#pragma unroll
        for (int t = 0; t < 4; ++t) {
            int ex  = seli[t];
            int pos = atomicAdd(&ecount[ex], 1);
            etok[(size_t)ex * CAP + pos]  = tok;
            egate[(size_t)ex * CAP + pos] = selv[t] / s;
        }
    }
}

// ---------------- Kernel 2: fused expert FFN (WMMA bf16) ----------------
__global__ __launch_bounds__(256) void moe_expert_kernel(
    const __bf16* __restrict__ xn,
    const float* __restrict__ W_in, const float* __restrict__ b_in,
    const float* __restrict__ W_out, const float* __restrict__ b_out,
    const int* __restrict__ ecount, const int* __restrict__ etok,
    const float* __restrict__ egate, float* __restrict__ out)
{
    const int e    = blockIdx.y;
    const int tile = blockIdx.x;
    const int cnt  = ecount[e];
    const int t0   = tile * MT;
    if (t0 >= cnt) return;                 // uniform: safe before barriers

    __shared__ __bf16 lds_x[MT * DDIM];    // 32 KB: x-hat token tile
    __shared__ __bf16 lds_h[MT * FFC];     // 16 KB: SiLU(h) chunk
    __shared__ int    s_tok[MT];
    __shared__ float  s_gate[MT];

    const int tid  = threadIdx.x;
    const int lane = tid & 31;
    const int wv   = tid >> 5;
    const int nl   = lane & 15;            // N column / A row
    const int kh   = lane >> 4;            // K half select
    const int khb  = kh * 16;              // A-frag K sub-offset
    const int kh16 = kh * 16;              // B-frag K sub-offset

    const float* We_in  = W_in  + (size_t)e * FF * DDIM;
    const float* We_out = W_out + (size_t)e * DDIM * FF;

    if (tid < MT) {
        int idx = t0 + tid;
        if (idx < cnt) { s_tok[tid] = etok[(size_t)e * CAP + idx];
                         s_gate[tid] = egate[(size_t)e * CAP + idx]; }
        else           { s_tok[tid] = etok[(size_t)e * CAP + t0];   // valid row, gate 0
                         s_gate[tid] = 0.f; }
    }
    __syncthreads();

    {   // stage x-hat tile via async global->LDS copy (16B/lane/issue).
        // 2048 uint4 slots total; thread t handles slots t + 256*i.
        const unsigned lds_base = (unsigned)(uintptr_t)lds_x;   // LDS aperture: addr[31:0]
#pragma unroll
        for (int i = 0; i < 8; ++i) {
            int s   = i * 256 + tid;
            int row = s >> 6;                      // 64 slots (1 KB) per row
            int col = (s & 63) * 8;                // bf16 units
            unsigned dst  = lds_base + (unsigned)s * 16u;
            uint64_t gsrc = (uint64_t)(uintptr_t)(xn + (size_t)s_tok[row] * DDIM + col);
            asm volatile("global_load_async_to_lds_b128 %0, %1, off"
                         :: "v"(dst), "v"(gsrc) : "memory");
        }
        asm volatile("s_wait_asynccnt 0" ::: "memory");
    }
    __syncthreads();

    v8f yacc[4][2];
#pragma unroll
    for (int j = 0; j < 4; ++j)
#pragma unroll
        for (int u = 0; u < 2; ++u)
            yacc[j][u] = (v8f){0.f,0.f,0.f,0.f,0.f,0.f,0.f,0.f};

    for (int fc = 0; fc < FF / FFC; ++fc) {
        const int ffbase = fc * FFC;

        // ---- GEMM1: h = x-hat @ W_in^T.  kk-outer: A frags loaded once,
        //      each B frag feeds 2 WMMAs (the two 16-row token sub-tiles).
        v8f hacc[2][2];
#pragma unroll
        for (int s = 0; s < 2; ++s)
#pragma unroll
            for (int u = 0; u < 2; ++u)
                hacc[s][u] = (v8f){0.f,0.f,0.f,0.f,0.f,0.f,0.f,0.f};

#pragma unroll 4
        for (int kk = 0; kk < DDIM / 32; ++kk) {
            v16bf a0 = load_a_frag(lds_x, DDIM, nl,      kk * 32 + khb);
            v16bf a1 = load_a_frag(lds_x, DDIM, nl + 16, kk * 32 + khb);
#pragma unroll
            for (int s = 0; s < 2; ++s) {
                const int ffcol = ffbase + (wv * 2 + s) * 16 + nl;
                const float* brow = We_in + (size_t)ffcol * DDIM + kk * 32 + kh16;
                __builtin_prefetch(brow + 64, 0, 1);
                v16bf b = load_b_frag_f32(brow);
                hacc[s][0] = __builtin_amdgcn_wmma_f32_16x16x32_bf16(
                                 false, a0, false, b, (short)0, hacc[s][0], false, false);
                hacc[s][1] = __builtin_amdgcn_wmma_f32_16x16x32_bf16(
                                 false, a1, false, b, (short)0, hacc[s][1], false, false);
            }
        }
        // bias + SiLU + bf16 store to LDS (C layout: m = r + 8*kh per sub-tile)
#pragma unroll
        for (int s = 0; s < 2; ++s) {
            const int ffl = (wv * 2 + s) * 16 + nl;
            const float bin = b_in[(size_t)e * FF + ffbase + ffl];
#pragma unroll
            for (int u = 0; u < 2; ++u)
#pragma unroll
                for (int r = 0; r < 8; ++r) {
                    float hv = hacc[s][u][r] + bin;
                    float sv = hv / (1.f + __expf(-hv));       // SiLU
                    lds_h[(u * 16 + r + kh * 8) * FFC + ffl] = (__bf16)sv;
                }
        }
        __syncthreads();

        // ---- GEMM2: y += h @ W_out^T.  kk-outer, 4 D-tiles x 2 sub-tiles.
#pragma unroll 4
        for (int kk = 0; kk < FFC / 32; ++kk) {
            v16bf a0 = load_a_frag(lds_h, FFC, nl,      kk * 32 + khb);
            v16bf a1 = load_a_frag(lds_h, FFC, nl + 16, kk * 32 + khb);
#pragma unroll
            for (int j = 0; j < 4; ++j) {
                const int dcol = (wv * 4 + j) * 16 + nl;
                const float* brow = We_out + (size_t)dcol * FF + ffbase + kk * 32 + kh16;
                __builtin_prefetch(brow + 64, 0, 1);
                v16bf b = load_b_frag_f32(brow);
                yacc[j][0] = __builtin_amdgcn_wmma_f32_16x16x32_bf16(
                                 false, a0, false, b, (short)0, yacc[j][0], false, false);
                yacc[j][1] = __builtin_amdgcn_wmma_f32_16x16x32_bf16(
                                 false, a1, false, b, (short)0, yacc[j][1], false, false);
            }
        }
        __syncthreads();
    }

    // ---- epilogue: add b_out, scale by gate, scatter-accumulate ----
#pragma unroll
    for (int j = 0; j < 4; ++j) {
        const int dcol = (wv * 4 + j) * 16 + nl;
        const float bo = b_out[(size_t)e * DDIM + dcol];
#pragma unroll
        for (int u = 0; u < 2; ++u)
#pragma unroll
            for (int r = 0; r < 8; ++r) {
                int m = u * 16 + r + kh * 8;
                float val = (yacc[j][u][r] + bo) * s_gate[m];
                atomicAdd(out + (size_t)s_tok[m] * DDIM + dcol, val);
            }
    }
}

extern "C" void kernel_launch(void* const* d_in, const int* in_sizes, int n_in,
                              void* d_out, int out_size, void* d_ws, size_t ws_size,
                              hipStream_t stream) {
    const float* x     = (const float*)d_in[0];
    const float* hexw  = (const float*)d_in[1];
    const float* W_in  = (const float*)d_in[2];
    const float* b_in  = (const float*)d_in[3];
    const float* W_out = (const float*)d_in[4];
    const float* b_out = (const float*)d_in[5];
    const float* gamma = (const float*)d_in[6];
    const float* beta  = (const float*)d_in[7];
    float* out = (float*)d_out;

    // workspace layout
    char* ws = (char*)d_ws;
    __bf16* xn  = (__bf16*)ws;                       // 2 MB
    size_t off  = (size_t)NTOK * DDIM * sizeof(__bf16);
    int*   ecount = (int*)(ws + off);  off += 256;   // 64 counters (padded)
    int*   etok   = (int*)(ws + off);  off += (size_t)NEXP * CAP * sizeof(int);
    float* egate  = (float*)(ws + off);

    hipMemsetAsync(d_out, 0, (size_t)out_size * sizeof(float), stream);
    hipMemsetAsync(ecount, 0, 256, stream);

    ln_topk_kernel<<<NTOK / 8, 256, 0, stream>>>(
        x, hexw, gamma, beta, xn, ecount, etok, egate);

    moe_expert_kernel<<<dim3(CAP / MT, NEXP), 256, 0, stream>>>(
        xn, W_in, b_in, W_out, b_out, ecount, etok, egate, out);
}
